// Decoder_47373489275469
// MI455X (gfx1250) — compile-verified, matrix operand order checked by reading, not run on hardware
//
#include <hip/hip_runtime.h>

// ---------------------------------------------------------------------------
// Problem constants (reference: B,S,E,H,T = 512,128,256,512,128)
// ---------------------------------------------------------------------------
#define BB 512
#define SS 128
#define EE 256
#define HH 512
#define TT 128

#define BTILE 16            // WMMA M
#define KG    768           // E + H   (fused [W_ih;W_hh] K)
#define NG    2048          // 4H
#define KTG   (KG/32)       // 24 k-tiles
#define KI    512
#define KTI   (KI/32)
#define KO    1024
#define KTO   (KO/32)

typedef __attribute__((ext_vector_type(16))) __bf16 v16bf;
typedef __attribute__((ext_vector_type(8)))  float  v8f;

__device__ __forceinline__ unsigned short f2bf(float f) {
  unsigned u = __builtin_bit_cast(unsigned, f);
  u = (u + 0x7FFFu + ((u >> 16) & 1u)) >> 16;     // RNE, NaN ignored (compile-only)
  return (unsigned short)u;
}

// --- gfx1250 hardware transcendentals --------------------------------------
// V_TANH_F32 is new on gfx1250 (TRANS32 op). Builtin if available, else asm
// with 2 trailing v_nop for the TRANS-op RAW hazard rule (ISA 15.5 / 7.4).
__device__ __forceinline__ float fast_tanh(float x) {
#if __has_builtin(__builtin_amdgcn_tanhf)
  return __builtin_amdgcn_tanhf(x);
#elif __has_builtin(__builtin_amdgcn_tanh_f32)
  return __builtin_amdgcn_tanh_f32(x);
#else
  float y;
  asm volatile("v_tanh_f32 %0, %1\n\tv_nop\n\tv_nop" : "=v"(y) : "v"(x));
  return y;
#endif
}
__device__ __forceinline__ float fast_exp2(float x) {
#if __has_builtin(__builtin_amdgcn_exp2f)
  return __builtin_amdgcn_exp2f(x);    // v_exp_f32
#else
  return exp2f(x);
#endif
}
__device__ __forceinline__ float fast_rcp(float x) {
#if __has_builtin(__builtin_amdgcn_rcpf)
  return __builtin_amdgcn_rcpf(x);     // v_rcp_f32
#else
  return 1.0f / x;
#endif
}
#define LOG2E 1.4426950408889634f
__device__ __forceinline__ float sigm(float x) {
  return fast_rcp(1.0f + fast_exp2(-LOG2E * x));
}

union FragU { v16bf v; uint4 u[2]; };

// A fragment (16x32 bf16) from an LDS row-major [16][K] bf16 panel.
// Lane layout (ISA 7.12.2): lane&15 = M; lanes>=16 hold K+8 group; VGPR4-7 = K+16.
__device__ __forceinline__ v16bf load_a_frag(const unsigned short* aPanel, int K, int kt, int lane) {
  int m  = lane & 15;
  int e0 = m * K + kt * 32 + ((lane >> 4) & 1) * 8;
  FragU r;
  r.u[0] = *(const uint4*)(aPanel + e0);
  r.u[1] = *(const uint4*)(aPanel + e0 + 16);
  return r.v;
}

// B fragment (32x16 bf16) from pre-swizzled global tiles:
// dst[(nt*KT+kt)*32 + lane] * 16 ushorts -> 32 contiguous bytes per lane.
__device__ __forceinline__ v16bf load_b_frag(const unsigned short* pk, int KT, int nt, int kt, int lane) {
  const uint4* p = (const uint4*)(pk + ((size_t)(nt * KT + kt) * 32 + lane) * 16);
  FragU r;
  r.u[0] = p[0];
  r.u[1] = p[1];
  return r.v;
}

__device__ __forceinline__ v8f wmma_row(const unsigned short* aPanel, int K, int KT,
                                        const unsigned short* pkB, int nt, int lane, v8f acc) {
#pragma unroll 4
  for (int kt = 0; kt < KT; ++kt) {
    v16bf a = load_a_frag(aPanel, K, kt, lane);
    v16bf b = load_b_frag(pkB, KT, nt, kt, lane);
    acc = __builtin_amdgcn_wmma_f32_16x16x32_bf16(false, a, false, b, (short)0, acc, false, false);
  }
  return acc;
}

// ---------------------------------------------------------------------------
// Pack a KxN row-major f32 matrix (optionally concat of two: rows [0,k0) from
// src0, rows [k0,K) from src1) into bf16 WMMA B-fragment tile layout.
// ---------------------------------------------------------------------------
__global__ void pack_b_tiles(const float* __restrict__ src0, const float* __restrict__ src1,
                             int k0, int K, int N, unsigned* __restrict__ dst) {
  int lane = threadIdx.x & 31, wav = threadIdx.x >> 5;
  int KT = K / 32, NT = N / 16;
  int tile = blockIdx.x * 8 + wav;
  if (tile >= KT * NT) return;
  int nt = tile / KT, kt = tile % KT;
  unsigned* out = dst + ((size_t)tile * 32 + lane) * 8;
  int col = nt * 16 + (lane & 15);
  int kb  = kt * 32 + ((lane >> 4) & 1) * 8;
#pragma unroll
  for (int j = 0; j < 8; ++j) {
    int k = kb + ((j >= 4) ? 16 : 0) + 2 * (j & 3);
    float f0 = (k     < k0) ? src0[(size_t)k * N + col]       : src1[(size_t)(k - k0) * N + col];
    float f1 = (k + 1 < k0) ? src0[(size_t)(k + 1) * N + col] : src1[(size_t)(k + 1 - k0) * N + col];
    out[j] = (unsigned)f2bf(f0) | ((unsigned)f2bf(f1) << 16);
  }
}

// ---------------------------------------------------------------------------
// ctx = context @ W_ctx + b_ctx   (loop-invariant attention context, f32 out)
// One block per 16 rows of the flattened (B*S, H) matrix; 8 waves x 4 n-tiles.
// ---------------------------------------------------------------------------
__global__ __launch_bounds__(256) void ctx_precompute(const float* __restrict__ context,
                                                      const unsigned short* __restrict__ pkCtx,
                                                      const float* __restrict__ b_ctx,
                                                      float* __restrict__ ctxOut) {
  __shared__ unsigned short aP[BTILE * HH];   // 16 KB bf16 A panel
  int tid = threadIdx.x, lane = tid & 31, wave = tid >> 5;
  size_t rowBase = (size_t)blockIdx.x * BTILE;

  for (int i = tid; i < BTILE * HH; i += 256)
    aP[i] = f2bf(context[rowBase * HH + i]);
  __syncthreads();

  int mBase = (lane >> 4) * 8;
#pragma unroll
  for (int nti = 0; nti < 4; ++nti) {
    int nt = wave * 4 + nti;
    int col = nt * 16 + (lane & 15);
    float bias = b_ctx[col];
    v8f acc;
#pragma unroll
    for (int r = 0; r < 8; ++r) acc[r] = bias;
    acc = wmma_row(aP, HH, KTI, pkCtx, nt, lane, acc);
#pragma unroll
    for (int r = 0; r < 8; ++r)
      ctxOut[(rowBase + r + mBase) * HH + col] = acc[r];
  }
}

// ---------------------------------------------------------------------------
// Persistent decoder: 32 workgroups x 16 batch rows, T=128 sequential steps.
// All recurrent state lives in LDS (~315 KB / WGP).
// ---------------------------------------------------------------------------
#define SMEM_BYTES (24576 + 32768 + 16384 + 131072 + 32768 + 32768 + 32768 + 8192 + 8192 + 2048 + 64)

__global__ __launch_bounds__(256) void decoder_persistent(
    const float* __restrict__ emb,        // (B,S,E)
    const float* __restrict__ dec0,       // (B,E)
    const float* __restrict__ h0,         // (B,H)
    const float* __restrict__ c0,         // (B,H)
    const float* __restrict__ ctx,        // (B,S,H) precomputed f32
    const unsigned short* __restrict__ pkG,
    const unsigned short* __restrict__ pkI,
    const unsigned short* __restrict__ pkO,
    const float* __restrict__ b_ih, const float* __restrict__ b_hh,
    const float* __restrict__ b_inp, const float* __restrict__ b_out,
    const float* __restrict__ Vw,
    float* __restrict__ outA,             // (B,T,S)
    float* __restrict__ outP,             // (B,T)
    float* __restrict__ outH,             // (B,H)
    float* __restrict__ outC) {           // (B,H)
  extern __shared__ unsigned char smem[];
  unsigned short* aG = (unsigned short*)smem;               // [16][768]
  unsigned short* aO = aG + BTILE * KG;                     // [16][1024]
  unsigned short* aI = aO + BTILE * KO;                     // [16][512]
  float* gates = (float*)(aI + BTILE * HH);                 // [16][2048]
  float* hbuf  = gates + BTILE * NG;
  float* cbuf  = hbuf + BTILE * HH;
  float* inp   = cbuf + BTILE * HH;
  float* attb  = inp + BTILE * HH;
  float* alph  = attb + BTILE * SS;
  float* vsh   = alph + BTILE * SS;
  int*   idxsh = (int*)(vsh + HH);

  const int tid = threadIdx.x, lane = tid & 31, wave = tid >> 5;
  const int b0 = blockIdx.x * BTILE;
  const int mBase = (lane >> 4) * 8;

  // ---- init carried state --------------------------------------------------
  for (int i = tid; i < HH; i += 256) vsh[i] = Vw[i];
  for (int i = tid; i < BTILE * HH; i += 256) {
    int b = i >> 9, hc = i & (HH - 1);
    float hv = h0[(size_t)(b0 + b) * HH + hc];
    hbuf[i] = hv;
    cbuf[i] = c0[(size_t)(b0 + b) * HH + hc];
    aG[b * KG + EE + hc] = f2bf(hv);
  }
  for (int i = tid; i < BTILE * EE; i += 256) {
    int b = i >> 8, e = i & (EE - 1);
    aG[b * KG + e] = f2bf(dec0[(size_t)(b0 + b) * EE + e]);
  }
  __syncthreads();

  for (int t = 0; t < TT; ++t) {
    // -- Phase 1: gates = [dec_in | h] @ [W_ih;W_hh] + b_ih + b_hh (WMMA bf16)
    for (int nti = 0; nti < 16; ++nti) {
      int nt = wave * 16 + nti;
      int col = nt * 16 + (lane & 15);
      float bias = b_ih[col] + b_hh[col];
      v8f acc;
#pragma unroll
      for (int r = 0; r < 8; ++r) acc[r] = bias;
      acc = wmma_row(aG, KG, KTG, pkG, nt, lane, acc);
#pragma unroll
      for (int r = 0; r < 8; ++r)
        gates[(r + mBase) * NG + col] = acc[r];
    }
    __syncthreads();

    // -- Phase 2: LSTM activations -> c_t, h_t (HW v_tanh_f32 / v_exp_f32)
    for (int i = tid; i < BTILE * HH; i += 256) {
      int b = i >> 9, hc = i & (HH - 1);
      const float* g = gates + b * NG;
      float ig = g[hc], fg = g[HH + hc], gg = g[2 * HH + hc], og = g[3 * HH + hc];
      float cn = sigm(fg) * cbuf[i] + sigm(ig) * fast_tanh(gg);
      float ht = sigm(og) * fast_tanh(cn);
      cbuf[i] = cn;
      hbuf[i] = ht;
      unsigned short hb = f2bf(ht);
      aI[b * KI + hc] = hb;
      aO[b * KO + HH + hc] = hb;
    }
    __syncthreads();

    // -- Phase 3: inp = h_t @ W_inp + b_inp (WMMA bf16)
    for (int nti = 0; nti < 4; ++nti) {
      int nt = wave * 4 + nti;
      int col = nt * 16 + (lane & 15);
      float bias = b_inp[col];
      v8f acc;
#pragma unroll
      for (int r = 0; r < 8; ++r) acc[r] = bias;
      acc = wmma_row(aI, KI, KTI, pkI, nt, lane, acc);
#pragma unroll
      for (int r = 0; r < 8; ++r)
        inp[(r + mBase) * HH + col] = acc[r];
    }
    __syncthreads();

    // -- Phase 4: att[b][s] = sum_h V[h]*tanh(inp[b][h]+ctx[b][s][h])
    //    Dominant phase: 1.05M tanh/step/WG -> single-instruction v_tanh_f32.
    for (int ri = 0; ri < 256; ++ri) {
      int row = wave * 256 + ri;
      int b = row >> 7, s = row & (SS - 1);
      const float4* c4 = (const float4*)(ctx + ((size_t)(b0 + b) * SS + s) * HH);
      const float4* i4 = (const float4*)(inp + b * HH);
      const float4* v4 = (const float4*)vsh;
      __builtin_prefetch(c4 + 128, 0, 0);   // next s row -> global_prefetch_b8
      float acc = 0.0f;
#pragma unroll
      for (int p = 0; p < 4; ++p) {
        float4 c = c4[p * 32 + lane], q = i4[p * 32 + lane], v = v4[p * 32 + lane];
        acc += v.x * fast_tanh(q.x + c.x) + v.y * fast_tanh(q.y + c.y) +
               v.z * fast_tanh(q.z + c.z) + v.w * fast_tanh(q.w + c.w);
      }
#pragma unroll
      for (int off = 16; off > 0; off >>= 1) acc += __shfl_xor(acc, off, 32);
      if (lane == 0) attb[b * SS + s] = acc;
    }
    __syncthreads();

    // -- Phase 5: per-row softmax + argmax; write alpha + pointer to d_out
    for (int rr = 0; rr < 2; ++rr) {
      int b = wave * 2 + rr;
      float e[4];
      float m = -3.4e38f;
      int mi = 0;
#pragma unroll
      for (int j = 0; j < 4; ++j) {
        float v = attb[b * SS + lane + j * 32];
        e[j] = v;
        if (v > m) { m = v; mi = lane + j * 32; }
      }
#pragma unroll
      for (int off = 16; off > 0; off >>= 1) {
        float om = __shfl_xor(m, off, 32);
        int omi = __shfl_xor(mi, off, 32);
        if (om > m || (om == m && omi < mi)) { m = om; mi = omi; }
      }
      float s = 0.0f;
#pragma unroll
      for (int j = 0; j < 4; ++j) { e[j] = fast_exp2(LOG2E * (e[j] - m)); s += e[j]; }
#pragma unroll
      for (int off = 16; off > 0; off >>= 1) s += __shfl_xor(s, off, 32);
      float inv = fast_rcp(s);
      size_t obase = ((size_t)(b0 + b) * TT + t) * SS;
#pragma unroll
      for (int j = 0; j < 4; ++j) {
        float a = e[j] * inv;
        alph[b * SS + lane + j * 32] = a;
        outA[obase + lane + j * 32] = a;
      }
      if (lane == 0) {
        idxsh[b] = mi;
        outP[(size_t)(b0 + b) * TT + t] = (float)mi;
      }
    }
    __syncthreads();

    // -- Phase 6: hid = ctx^T alpha  -> first half of aO panel (bf16)
    for (int i = tid; i < BTILE * HH; i += 256) {
      int b = i >> 9, hc = i & (HH - 1);
      const float* cb = ctx + (size_t)(b0 + b) * SS * HH + hc;
      const float* al = alph + b * SS;
      float acc = 0.0f;
#pragma unroll 4
      for (int s = 0; s < SS; ++s) acc += al[s] * cb[(size_t)s * HH];
      aO[b * KO + hc] = f2bf(acc);
    }
    __syncthreads();

    // -- Phase 7: h_new = tanh([hid|h_t] @ W_out + b_out) (WMMA bf16)
    for (int nti = 0; nti < 4; ++nti) {
      int nt = wave * 4 + nti;
      int col = nt * 16 + (lane & 15);
      float bias = b_out[col];
      v8f acc;
#pragma unroll
      for (int r = 0; r < 8; ++r) acc[r] = bias;
      acc = wmma_row(aO, KO, KTO, pkO, nt, lane, acc);
#pragma unroll
      for (int r = 0; r < 8; ++r) {
        float hn = fast_tanh(acc[r]);
        int m = r + mBase;
        hbuf[m * HH + col] = hn;
        aG[m * KG + EE + col] = f2bf(hn);     // next-step LSTM h input
      }
    }
    __syncthreads();

    // -- Phase 8: dec_next = embedded_inputs[b, argmax, :] -> aG panel (bf16)
    for (int i = tid; i < BTILE * EE; i += 256) {
      int b = i >> 8, e = i & (EE - 1);
      int id = idxsh[b];
      aG[b * KG + e] = f2bf(emb[((size_t)(b0 + b) * SS + id) * EE + e]);
    }
    __syncthreads();
  }

  // ---- final h_last / c_last ----------------------------------------------
  for (int i = tid; i < BTILE * HH; i += 256) {
    int b = i >> 9, hc = i & (HH - 1);
    outH[(size_t)(b0 + b) * HH + hc] = hbuf[i];
    outC[(size_t)(b0 + b) * HH + hc] = cbuf[i];
  }
}

// ---------------------------------------------------------------------------
// Launch
// ---------------------------------------------------------------------------
extern "C" void kernel_launch(void* const* d_in, const int* in_sizes, int n_in,
                              void* d_out, int out_size, void* d_ws, size_t ws_size,
                              hipStream_t stream) {
  const float* emb   = (const float*)d_in[0];
  const float* dec0  = (const float*)d_in[1];
  const float* h0    = (const float*)d_in[2];
  const float* c0    = (const float*)d_in[3];
  const float* cont  = (const float*)d_in[4];
  const float* W_ih  = (const float*)d_in[5];
  const float* b_ih  = (const float*)d_in[6];
  const float* W_hh  = (const float*)d_in[7];
  const float* b_hh  = (const float*)d_in[8];
  const float* W_out = (const float*)d_in[9];
  const float* b_out = (const float*)d_in[10];
  const float* W_inp = (const float*)d_in[11];
  const float* b_inp = (const float*)d_in[12];
  const float* W_ctx = (const float*)d_in[13];
  const float* b_ctx = (const float*)d_in[14];
  const float* Vw    = (const float*)d_in[15];

  // workspace layout (bytes)
  unsigned char* ws = (unsigned char*)d_ws;
  unsigned short* pkG   = (unsigned short*)(ws);                 // 768x2048 bf16: 3,145,728
  unsigned short* pkI   = (unsigned short*)(ws + 3145728);       // 512x512  bf16:   524,288
  unsigned short* pkO   = (unsigned short*)(ws + 3670016);       // 1024x512 bf16: 1,048,576
  unsigned short* pkCtx = (unsigned short*)(ws + 4718592);       // 512x512  bf16:   524,288
  float*          ctxF  = (float*)(ws + 5242880);                // (B,S,H) f32: 134,217,728

  float* outA = (float*)d_out;
  float* outP = outA + (size_t)BB * TT * SS;
  float* outH = outP + (size_t)BB * TT;
  float* outC = outH + (size_t)BB * HH;

  // 1) pack weights into WMMA B-fragment tile layout (bf16)
  {
    int tiles = (NG / 16) * KTG;   // 3072
    pack_b_tiles<<<(tiles + 7) / 8, 256, 0, stream>>>(W_ih, W_hh, EE, KG, NG, (unsigned*)pkG);
  }
  {
    int tiles = (HH / 16) * KTI;   // 512
    pack_b_tiles<<<(tiles + 7) / 8, 256, 0, stream>>>(W_inp, W_inp, KI, KI, HH, (unsigned*)pkI);
  }
  {
    int tiles = (HH / 16) * KTO;   // 1024
    pack_b_tiles<<<(tiles + 7) / 8, 256, 0, stream>>>(W_out, W_out, KO, KO, HH, (unsigned*)pkO);
  }
  {
    int tiles = (HH / 16) * KTI;   // 512
    pack_b_tiles<<<(tiles + 7) / 8, 256, 0, stream>>>(W_ctx, W_ctx, KI, KI, HH, (unsigned*)pkCtx);
  }

  // 2) loop-invariant attention context (WMMA GEMM, f32 out -> L2 resident)
  ctx_precompute<<<(BB * SS) / BTILE, 256, 0, stream>>>(cont, pkCtx, b_ctx, ctxF);

  // 3) persistent decoder: 32 WGs x 16 batch rows x 128 steps, ~315 KB LDS each
  (void)hipFuncSetAttribute((const void*)decoder_persistent,
                            hipFuncAttributeMaxDynamicSharedMemorySize, SMEM_BYTES);
  decoder_persistent<<<BB / BTILE, 256, SMEM_BYTES, stream>>>(
      emb, dec0, h0, c0, ctxF, pkG, pkI, pkO,
      b_ih, b_hh, b_inp, b_out, Vw, outA, outP, outH, outC);
}
